// knrm_23940147708478
// MI455X (gfx1250) — matrix operand order repaired and checked by reading, not compiled, not run on hardware
//
#include <hip/hip_runtime.h>
#include <hip/hip_bf16.h>

typedef __attribute__((ext_vector_type(16))) _Float16 v16h;
typedef __attribute__((ext_vector_type(8)))  float    v8f;

#define EDIM 300
#define EPAD 320
#define NB   128
#define NLQ  64
#define NLD  512

union HVec { float4 f[2]; v16h h; };

// ---------------------------------------------------------------------------
// Kernel 1: per-row L2 normalize (eps semantics match torch F.normalize) and
// convert to f16, padding E from 300 to 320 with zeros. One wave32 per row.
// ---------------------------------------------------------------------------
__global__ void knrm_norm_f16_kernel(const float* __restrict__ src,
                                     _Float16* __restrict__ dst, int nrows) {
  int row  = blockIdx.x * 4 + (threadIdx.x >> 5);
  int lane = threadIdx.x & 31;
  if (row >= nrows) return;
  const float* s = src + (size_t)row * EDIM;
  float acc = 0.0f;
  for (int i = lane; i < EDIM; i += 32) { float x = s[i]; acc += x * x; }
#pragma unroll
  for (int m = 16; m >= 1; m >>= 1) acc += __shfl_xor(acc, m, 32);
  float inv = 1.0f / fmaxf(sqrtf(acc), 1e-12f);
  _Float16* d = dst + (size_t)row * EPAD;
  for (int i = lane; i < EPAD; i += 32) {
    float v = (i < EDIM) ? s[i] * inv : 0.0f;
    d[i] = (_Float16)v;
  }
}

// ---------------------------------------------------------------------------
// Kernel 2: fused cosine-GEMM (WMMA f16 -> f32) + RBF kernel pooling.
// Block = (batch, q-tile of 16 rows), 4 waves; each wave owns 8 doc-tiles.
// Two independent WMMA accumulator chains (even/odd K-step) so B-loads for
// step k+1 can be in flight while step k's WMMA executes.
// ---------------------------------------------------------------------------
__global__ void __launch_bounds__(128)
knrm_pool_kernel(const _Float16* __restrict__ qh,
                 const _Float16* __restrict__ dh,
                 const float* __restrict__ mask_q,
                 const float* __restrict__ mask_d,
                 float* __restrict__ feats_part) {
  // K-NRM kernel centers / -1/(2*sigma^2) coefficients (K=11)
  const float MU[11] = {1.0f, 0.9f, 0.7f, 0.5f, 0.3f, 0.1f,
                        -0.1f, -0.3f, -0.5f, -0.7f, -0.9f};
  const float CF[11] = {-500000.0f, -50.0f, -50.0f, -50.0f, -50.0f, -50.0f,
                        -50.0f, -50.0f, -50.0f, -50.0f, -50.0f};

  const int b    = blockIdx.x >> 2;   // batch
  const int qt   = blockIdx.x & 3;    // q-tile (16 rows each)
  const int tid  = threadIdx.x;
  const int wave = tid >> 5;
  const int lane = tid & 31;
  const int grp  = lane >> 4;         // half-wave select
  const int lm   = lane & 15;         // M row (A) / N col (B)

  __shared__ __align__(16) _Float16 sA[16 * EPAD];   // q-tile, 10 KB
  __shared__ float sPool[4][16][11];                 // per-wave partial sums

  // Stage the 16x320 f16 q-tile into LDS (contiguous 10240B block).
  {
    const uint4* s = (const uint4*)(qh + ((size_t)b * NLQ + qt * 16) * EPAD);
    uint4* d = (uint4*)sA;
#pragma unroll
    for (int i = 0; i < 5; ++i) d[tid + 128 * i] = s[tid + 128 * i];
  }
  __syncthreads();

  float pool[8][11];
#pragma unroll
  for (int r = 0; r < 8; ++r)
#pragma unroll
    for (int k = 0; k < 11; ++k) pool[r][k] = 0.0f;

  for (int j = 0; j < 8; ++j) {
    const int nt = wave + 4 * j;                     // doc tile index (0..31)
    v8f acc0 = {};                                   // even K-steps
    v8f acc1 = {};                                   // odd  K-steps
    // B fragment base: doc rows are K-contiguous == WMMA column-major B layout.
    const _Float16* brow =
        dh + ((size_t)b * NLD + nt * 16 + lm) * EPAD + grp * 8;
#pragma unroll
    for (int kk = 0; kk < 10; kk += 2) {             // 10 x K=32 covers EPAD=320
      HVec av0, bv0, av1, bv1;
      const _Float16* pa0 = sA + lm * EPAD + kk * 32 + grp * 8;
      av0.f[0] = *(const float4*)pa0;                // K chunk   (grp*8 .. +7)
      av0.f[1] = *(const float4*)(pa0 + 16);         // K chunk +16
      const _Float16* pb0 = brow + kk * 32;
      bv0.f[0] = *(const float4*)pb0;                // K chunk (grp*8 .. +7), then +16
      bv0.f[1] = *(const float4*)(pb0 + 16);
      const _Float16* pa1 = pa0 + 32;
      av1.f[0] = *(const float4*)pa1;
      av1.f[1] = *(const float4*)(pa1 + 16);
      const _Float16* pb1 = pb0 + 32;
      bv1.f[0] = *(const float4*)pb1;
      bv1.f[1] = *(const float4*)(pb1 + 16);
      acc0 = __builtin_amdgcn_wmma_f32_16x16x32_f16(
          false, av0.h, false, bv0.h, (short)0, acc0, false, false);
      acc1 = __builtin_amdgcn_wmma_f32_16x16x32_f16(
          false, av1.h, false, bv1.h, (short)0, acc1, false, false);
    }
    // RBF pooling on the 16x16 sim tile; fold in mask_d for this doc column.
    const float md = mask_d[b * NLD + nt * 16 + lm];
#pragma unroll
    for (int r = 0; r < 8; ++r) {
      const float sv = acc0[r] + acc1[r];
#pragma unroll
      for (int k = 0; k < 11; ++k) {
        const float df = sv - MU[k];
        pool[r][k] += md * __expf(df * df * CF[k]);
      }
    }
  }

  // Reduce over the doc (N) dimension: N lives across the 16-lane halves.
#pragma unroll
  for (int r = 0; r < 8; ++r)
#pragma unroll
    for (int k = 0; k < 11; ++k) {
      float v = pool[r][k];
      v += __shfl_xor(v, 1, 32);
      v += __shfl_xor(v, 2, 32);
      v += __shfl_xor(v, 4, 32);
      v += __shfl_xor(v, 8, 32);
      pool[r][k] = v;
    }
  if (lm == 0) {
#pragma unroll
    for (int r = 0; r < 8; ++r)
#pragma unroll
      for (int k = 0; k < 11; ++k)
        sPool[wave][grp * 8 + r][k] = pool[r][k];
  }
  __syncthreads();

  // log-pool per q row, apply mask_q, sum over the 16 q rows of this tile.
  if (tid < 11) {
    const int k = tid;
    float part = 0.0f;
    for (int m = 0; m < 16; ++m) {
      float ps = sPool[0][m][k] + sPool[1][m][k] + sPool[2][m][k] + sPool[3][m][k];
      float lp = logf(fmaxf(ps, 1e-10f)) * 0.01f * mask_q[b * NLQ + qt * 16 + m];
      part += lp;
    }
    feats_part[(size_t)blockIdx.x * 11 + k] = part;   // [b][qt][k]
  }
}

// ---------------------------------------------------------------------------
// Kernel 3: sum the 4 q-tile partials, tiny dense layer + tanh.
// ---------------------------------------------------------------------------
__global__ void knrm_final_kernel(const float* __restrict__ feats_part,
                                  const float* __restrict__ w,
                                  const float* __restrict__ bias,
                                  float* __restrict__ out) {
  int b = blockIdx.x * blockDim.x + threadIdx.x;
  if (b >= NB) return;
  float f = 0.0f;
#pragma unroll
  for (int k = 0; k < 11; ++k) {
    float s = feats_part[(size_t)(b * 4 + 0) * 11 + k] +
              feats_part[(size_t)(b * 4 + 1) * 11 + k] +
              feats_part[(size_t)(b * 4 + 2) * 11 + k] +
              feats_part[(size_t)(b * 4 + 3) * 11 + k];
    f += s * w[k];
  }
  out[b] = tanhf(f + bias[0]);
}

// ---------------------------------------------------------------------------
extern "C" void kernel_launch(void* const* d_in, const int* in_sizes, int n_in,
                              void* d_out, int out_size, void* d_ws, size_t ws_size,
                              hipStream_t stream) {
  const float* q  = (const float*)d_in[0];  // [128,64,300]
  const float* dd = (const float*)d_in[1];  // [128,512,300]
  const float* mq = (const float*)d_in[2];  // [128,64]
  const float* md = (const float*)d_in[3];  // [128,512]
  const float* w  = (const float*)d_in[4];  // [1,11]
  const float* bb = (const float*)d_in[5];  // [1]

  char* ws = (char*)d_ws;
  const size_t qh_bytes = (size_t)NB * NLQ * EPAD * sizeof(_Float16); // 5.24 MB
  const size_t dh_bytes = (size_t)NB * NLD * EPAD * sizeof(_Float16); // 41.9 MB
  _Float16* qh = (_Float16*)ws;
  _Float16* dh = (_Float16*)(ws + qh_bytes);
  float* feats_part = (float*)(ws + qh_bytes + dh_bytes);             // [128][4][11]

  knrm_norm_f16_kernel<<<(NB * NLQ) / 4, 128, 0, stream>>>(q, qh, NB * NLQ);
  knrm_norm_f16_kernel<<<(NB * NLD) / 4, 128, 0, stream>>>(dd, dh, NB * NLD);
  knrm_pool_kernel<<<NB * 4, 128, 0, stream>>>(qh, dh, mq, md, feats_part);
  knrm_final_kernel<<<1, 128, 0, stream>>>(feats_part, w, bb, (float*)d_out);
}
